// Generator_89206470738255
// MI455X (gfx1250) — matrix operand order repaired and checked by reading, not compiled, not run on hardware
//
#include <hip/hip_runtime.h>

#define LATENT   128
#define HIDDEN   512
#define BATCH    256
#define IN_DIM   5
#define T_STEPS  256
#define GATES    2048   // 4*HIDDEN

typedef __attribute__((ext_vector_type(16))) _Float16 v16h;
typedef __attribute__((ext_vector_type(8)))  _Float16 v8h;
typedef __attribute__((ext_vector_type(8)))  float    v8f;

__device__ __forceinline__ float sigmoidf_(float x) {
    return 1.0f / (1.0f + __expf(-x));
}

// ---------------------------------------------------------------------------
// Pack w_hh0 | w_ih1 | w_hh1 (each [512,2048] f32 row-major) into f16 WMMA
// B-tile layout: tiles indexed [kt(0..15)][nt(0..127)], 512 f16 per tile,
// lane holds N = nt*16 + (lane&15), K = kt*32 + (lane>>4)*8 + (e&7) + (e>>3)*16,
// e=0..15 contiguous per lane (-> two global_load_b128 at runtime).
// ---------------------------------------------------------------------------
__global__ void pack_weights(const float* __restrict__ w_hh0,
                             const float* __restrict__ w_ih1,
                             const float* __restrict__ w_hh1,
                             _Float16* __restrict__ dst) {
    const long MAT = (long)HIDDEN * GATES; // 1,048,576
    for (long d = (long)blockIdx.x * blockDim.x + threadIdx.x;
         d < 3 * MAT;
         d += (long)gridDim.x * blockDim.x) {
        int  mat    = (int)(d / MAT);
        int  r      = (int)(d % MAT);
        int  tile   = r >> 9;        // /512
        int  within = r & 511;
        int  lane   = within >> 4;
        int  e      = within & 15;
        int  kt     = tile >> 7;     // 128 n-tiles per k-tile row
        int  nt     = tile & 127;
        int  K      = kt * 32 + (lane >> 4) * 8 + (e & 7) + (e >> 3) * 16;
        int  N      = nt * 16 + (lane & 15);
        const float* src = (mat == 0) ? w_hh0 : (mat == 1 ? w_ih1 : w_hh1);
        dst[d] = (_Float16)src[(long)K * GATES + N];
    }
}

// ---------------------------------------------------------------------------
// hidden = (z @ w_lat + b_lat).reshape(2, B, H) row-major (torch .view
// semantics): reshaped flat index maps to [B,2H] element (flat>>10, flat&1023).
// Also zeros c0/c1 and sets the stroke-5 start token x0[:,3]=1.
// ---------------------------------------------------------------------------
__global__ void init_state(const float* __restrict__ z,
                           const float* __restrict__ w_lat,
                           const float* __restrict__ b_lat,
                           _Float16* __restrict__ h0, _Float16* __restrict__ h1,
                           float* __restrict__ c0, float* __restrict__ c1,
                           float* __restrict__ xbuf) {
    long idx = (long)blockIdx.x * blockDim.x + threadIdx.x;
    if (idx < 2L * BATCH * HIDDEN) {
        int  l    = (int)(idx >> 17);       // 131072 = 256*512
        int  bsrc = (int)(idx >> 10);       // row in [B, 2H]
        int  csrc = (int)(idx & 1023);      // col in [B, 2H]
        float acc = b_lat[csrc];
        for (int k = 0; k < LATENT; ++k)
            acc += z[bsrc * LATENT + k] * w_lat[k * (2 * HIDDEN) + csrc];
        int b2 = (int)((idx >> 9) & 255);
        int j  = (int)(idx & 511);
        if (l == 0) { h0[b2 * HIDDEN + j] = (_Float16)acc; c0[b2 * HIDDEN + j] = 0.0f; }
        else        { h1[b2 * HIDDEN + j] = (_Float16)acc; c1[b2 * HIDDEN + j] = 0.0f; }
    }
    if (idx < BATCH * IN_DIM)
        xbuf[idx] = ((idx % IN_DIM) == 3) ? 1.0f : 0.0f;
}

// ---------------------------------------------------------------------------
// One LSTM layer step for the whole batch.
// gates[B,4H] = (x @ w_ih_x if x) + A1 @ B1pack + (A2 @ B2pack if A2) + bias
// Grid: 64 blocks x 256 threads = 512 waves. wave wid = m*32 + n:
//   m = output row tile (16 rows), n = hidden-unit tile (16 units).
// Each wave computes the 4 gate tiles for its hidden units, then does the
// cell update in registers. K-loop is software-pipelined (double-buffered
// A/B tiles) so next-tile L2 loads overlap the WMMAs: with ~1 wave/SIMD
// there is no occupancy to hide load latency otherwise.
// ---------------------------------------------------------------------------
__global__ void lstm_layer(const _Float16* __restrict__ A1, const _Float16* __restrict__ B1,
                           const _Float16* __restrict__ A2, const _Float16* __restrict__ B2,
                           const float* __restrict__ x,     const float* __restrict__ w_ih_x,
                           const float* __restrict__ bias,
                           float* __restrict__ c, _Float16* __restrict__ h_out) {
    const int lane   = threadIdx.x & 31;
    const int wid    = blockIdx.x * (blockDim.x >> 5) + (threadIdx.x >> 5);
    const int m      = wid >> 5;     // 0..15
    const int n      = wid & 31;     // 0..31
    const int laneM  = lane & 15;
    const int laneHi = lane >> 4;

    // Accumulator init: bias (+ tiny K=5 input GEMM for layer 0).
    // C/D layout: element acc[g][r] is row m*16 + r + laneHi*8, col n*16 + laneM.
    v8f acc[4];
#pragma unroll
    for (int g = 0; g < 4; ++g) {
        const int gcol = g * HIDDEN + n * 16 + laneM;
#pragma unroll
        for (int r = 0; r < 8; ++r) {
            const int row = m * 16 + r + laneHi * 8;
            float v = bias[gcol];
            if (x) {
#pragma unroll
                for (int k = 0; k < IN_DIM; ++k)
                    v += x[row * IN_DIM + k] * w_ih_x[k * GATES + gcol];
            }
            acc[g][r] = v;
        }
    }

    // A-tile (16-bit 16x32 layout): lane L holds M = L&15,
    // K in {kb..kb+7} U {16+kb..16+kb+7}, kb = (L>>4)*8.
    const int rowA = m * 16 + laneM;
    const int kb   = laneHi * 8;

    auto loadA = [&](const _Float16* __restrict__ A, int kt) -> v16h {
        const _Float16* ap = A + (long)rowA * HIDDEN + kt * 32;
        v8h lo = *(const v8h*)(ap + kb);
        v8h hi = *(const v8h*)(ap + 16 + kb);
        v16h a;
#pragma unroll
        for (int i = 0; i < 8; ++i) { a[i] = lo[i]; a[i + 8] = hi[i]; }
        return a;
    };
    auto loadB = [&](const _Float16* __restrict__ B, int kt, int g) -> v16h {
        const _Float16* bp = B + ((long)(kt * 128 + g * 32 + n) * 512 + lane * 16);
        v8h lo = *(const v8h*)bp;
        v8h hi = *(const v8h*)(bp + 8);
        v16h b;
#pragma unroll
        for (int i = 0; i < 8; ++i) { b[i] = lo[i]; b[i + 8] = hi[i]; }
        return b;
    };

    // K=512 GEMM pass, software-pipelined: prefetch k-tile kt+1 while the
    // WMMAs consume k-tile kt. Unrolled by 2 so buffer indices are constants
    // (no register rotation). Last iteration wraps its prefetch to tile 0
    // (harmless redundant loads) to keep loads unconditional / EXEC uniform.
    auto gemm_pass = [&](const _Float16* __restrict__ A, const _Float16* __restrict__ B) {
        v16h aBuf[2];
        v16h bBuf[2][4];
        aBuf[0] = loadA(A, 0);
#pragma unroll
        for (int g = 0; g < 4; ++g) bBuf[0][g] = loadB(B, 0, g);
#pragma unroll 2
        for (int kt = 0; kt < 16; ++kt) {
            const int cur = kt & 1, nxt = cur ^ 1;
            const int ktp = (kt + 1) & 15;
            aBuf[nxt] = loadA(A, ktp);
#pragma unroll
            for (int g = 0; g < 4; ++g) bBuf[nxt][g] = loadB(B, ktp, g);
#pragma unroll
            for (int g = 0; g < 4; ++g)
                acc[g] = __builtin_amdgcn_wmma_f32_16x16x32_f16(
                    false, aBuf[cur], false, bBuf[cur][g], (short)0, acc[g], false, false);
        }
    };
    gemm_pass(A1, B1);
    if (A2) gemm_pass(A2, B2);

    // Cell update: i,f,g,o for identical (row, hidden-unit) live in the same
    // register slot across the four accumulators.
#pragma unroll
    for (int r = 0; r < 8; ++r) {
        const int row = m * 16 + r + laneHi * 8;
        const int col = n * 16 + laneM;
        float iv = sigmoidf_(acc[0][r]);
        float fv = sigmoidf_(acc[1][r]);
        float gv = tanhf(acc[2][r]);
        float ov = sigmoidf_(acc[3][r]);
        float cold = c[row * HIDDEN + col];
        float cnew = fv * cold + iv * gv;
        float hnew = ov * tanhf(cnew);
        c[row * HIDDEN + col]     = cnew;
        h_out[row * HIDDEN + col] = (_Float16)hnew;
    }
}

// ---------------------------------------------------------------------------
// y = h1 @ w_out + b_out  ([256,5]); feeds back as next x and stores d_out[b,t,:].
// ---------------------------------------------------------------------------
__global__ void out_proj(const _Float16* __restrict__ h1,
                         const float* __restrict__ w_out, const float* __restrict__ b_out,
                         float* __restrict__ xbuf, float* __restrict__ out, int t) {
    int idx = blockIdx.x * blockDim.x + threadIdx.x;
    if (idx >= BATCH * IN_DIM) return;
    int b = idx / IN_DIM, d = idx % IN_DIM;
    float acc = b_out[d];
    for (int k = 0; k < HIDDEN; ++k)
        acc += (float)h1[b * HIDDEN + k] * w_out[k * IN_DIM + d];
    xbuf[idx] = acc;
    out[(long)b * T_STEPS * IN_DIM + (long)t * IN_DIM + d] = acc;
}

// ---------------------------------------------------------------------------
extern "C" void kernel_launch(void* const* d_in, const int* in_sizes, int n_in,
                              void* d_out, int out_size, void* d_ws, size_t ws_size,
                              hipStream_t stream) {
    const float* z     = (const float*)d_in[0];
    const float* w_lat = (const float*)d_in[1];
    const float* b_lat = (const float*)d_in[2];
    const float* w_ih0 = (const float*)d_in[3];
    const float* w_hh0 = (const float*)d_in[4];
    const float* b0    = (const float*)d_in[5];
    const float* w_ih1 = (const float*)d_in[6];
    const float* w_hh1 = (const float*)d_in[7];
    const float* b1    = (const float*)d_in[8];
    const float* w_out = (const float*)d_in[9];
    const float* b_out = (const float*)d_in[10];
    float* out = (float*)d_out;

    char*  ws  = (char*)d_ws;
    size_t off = 0;
    auto walloc = [&](size_t bytes) -> void* {
        void* p = ws + off;
        off = (off + bytes + 255) & ~(size_t)255;
        return p;
    };
    _Float16* packB = (_Float16*)walloc(3UL * HIDDEN * GATES * sizeof(_Float16));
    _Float16* H0[2] = { (_Float16*)walloc(BATCH * HIDDEN * sizeof(_Float16)),
                        (_Float16*)walloc(BATCH * HIDDEN * sizeof(_Float16)) };
    _Float16* H1[2] = { (_Float16*)walloc(BATCH * HIDDEN * sizeof(_Float16)),
                        (_Float16*)walloc(BATCH * HIDDEN * sizeof(_Float16)) };
    float* C0   = (float*)walloc(BATCH * HIDDEN * sizeof(float));
    float* C1   = (float*)walloc(BATCH * HIDDEN * sizeof(float));
    float* xbuf = (float*)walloc(BATCH * IN_DIM * sizeof(float));

    pack_weights<<<2048, 256, 0, stream>>>(w_hh0, w_ih1, w_hh1, packB);
    init_state<<<1024, 256, 0, stream>>>(z, w_lat, b_lat, H0[0], H1[0], C0, C1, xbuf);

    const _Float16* pHH0 = packB;
    const _Float16* pIH1 = packB + (size_t)HIDDEN * GATES;
    const _Float16* pHH1 = packB + 2 * (size_t)HIDDEN * GATES;

    for (int t = 0; t < T_STEPS; ++t) {
        const int rd = t & 1, wr = rd ^ 1;
        // layer 0: gates = x @ w_ih0 + h0 @ w_hh0 + b0
        lstm_layer<<<64, 256, 0, stream>>>(H0[rd], pHH0, nullptr, nullptr,
                                           xbuf, w_ih0, b0, C0, H0[wr]);
        // layer 1: gates = h0_new @ w_ih1 + h1 @ w_hh1 + b1
        lstm_layer<<<64, 256, 0, stream>>>(H0[wr], pIH1, H1[rd], pHH1,
                                           nullptr, nullptr, b1, C1, H1[wr]);
        // y = h1_new @ w_out + b_out  -> feedback x and d_out[:, t, :]
        out_proj<<<5, 256, 0, stream>>>(H1[wr], w_out, b_out, xbuf, out, t);
    }
}
